// LSTM_55027120996587
// MI455X (gfx1250) — matrix operand order repaired and checked by reading, not compiled, not run on hardware
//
#include <hip/hip_runtime.h>
#include <cstdint>
#include <cstddef>

// Problem constants (from reference): B=64, T=512, SIZE=512, H=1024, L=2, OUT=512.
// Only batch row 63 contributes to the output (ys[:, -1, :] selects batch 63).
#define HDIM   1024
#define TSTEPS 512
#define INSZ   512
#define OUTSZ  512
#define GATES  4096          // 4*H
#define NWG    64            // workgroups in persistent LSTM kernel
#define ZCOLS  64            // z-columns per WG (= 16 h-cols * 4 gates)
#define WPAD   1032          // padded LDS row stride (u16 elems) to rotate banks

typedef __bf16 bf16x16 __attribute__((ext_vector_type(16)));
typedef float  f32x8   __attribute__((ext_vector_type(8)));

union BFrag { bf16x16 v; uint4 q[2]; };

__device__ __forceinline__ float bflo(uint32_t u) { return __builtin_bit_cast(float, u << 16); }
__device__ __forceinline__ float bfhi(uint32_t u) { return __builtin_bit_cast(float, u & 0xFFFF0000u); }
__device__ __forceinline__ uint16_t f2bf(float f) {
  uint32_t u = __builtin_bit_cast(uint32_t, f);
  u += 0x7FFFu + ((u >> 16) & 1u);               // round-to-nearest-even
  return (uint16_t)(u >> 16);
}
__device__ __forceinline__ float sigmoidf(float x) { return 1.0f / (1.0f + __expf(-x)); }

// ---------------------------------------------------------------------------
// f32 -> bf16 cast with transpose: src is R x C row-major, dst is C x R row-major.
__global__ void cast_transpose(const float* __restrict__ src, uint16_t* __restrict__ dst,
                               int R, int C) {
  int i = blockIdx.x * 256 + threadIdx.x;
  if (i < R * C) {
    int r = i / C, c = i % C;
    dst[(size_t)c * R + r] = f2bf(src[i]);
  }
}

// Straight f32 -> bf16 cast.
__global__ void cast_copy(const float* __restrict__ src, uint16_t* __restrict__ dst, int n) {
  int i = blockIdx.x * 256 + threadIdx.x;
  if (i < n) dst[i] = f2bf(src[i]);
}

// Init: h0/h1 rows of the history buffers from state[63], zero barrier counters.
__global__ void init_misc(const float* __restrict__ state63,
                          uint16_t* __restrict__ h0row, uint16_t* __restrict__ h1row,
                          unsigned* __restrict__ ctrs) {
  int i = blockIdx.x * 256 + threadIdx.x;
  if (i < HDIM) {
    h0row[i] = f2bf(state63[HDIM + i]);        // h0 = state[63, H:2H]
    h1row[i] = f2bf(state63[3 * HDIM + i]);    // h1 = state[63, 3H:4H]
  }
  if (i < 4) ctrs[i] = 0u;
}

// ---------------------------------------------------------------------------
// bf16 WMMA GEMM: C(MxN,f32,row-major) = A(MxK,bf16,row-major) @ Bt^T + bias
// Bt is the transposed B: (N x K) row-major bf16, so each lane's K-run is contiguous.
// One wave computes one 16x16 tile via v_wmma_f32_16x16x32_bf16.
__global__ __launch_bounds__(128) void gemm_bf16(
    const uint16_t* __restrict__ A, int lda,
    const uint16_t* __restrict__ Bt, int ldb,
    const float* __restrict__ bias,
    float* __restrict__ C, int ldc,
    int Mt, int Nt, int K) {
  int wid  = blockIdx.x * 4 + (threadIdx.x >> 5);
  int lane = threadIdx.x & 31;
  if (wid >= Mt * Nt) return;
  int tm = wid / Nt, tn = wid % Nt;

  // A fragment (16x32 bf16): lane L holds row M=L%16, K = kA..kA+7 and kA+16..kA+23,
  // kA = 8*(L/16)  (ISA 16-bit A-matrix layout).
  int mrow = tm * 16 + (lane & 15);
  int kA   = (lane >> 4) * 8;
  // B fragment (32x16 bf16): lane L holds col N=L%16, K = kB..kB+7 and kB+16..kB+23, kB = 8*(L/16).
  int nrow = tn * 16 + (lane & 15);
  int kB   = (lane >> 4) * 8;

  const uint16_t* ap = A  + (size_t)mrow * lda + kA;
  const uint16_t* bp = Bt + (size_t)nrow * ldb + kB;

  f32x8 acc = {};
  for (int k0 = 0; k0 < K; k0 += 32) {
    BFrag a, b;
    a.q[0] = *(const uint4*)(ap + k0);         // K = kA .. kA+7
    a.q[1] = *(const uint4*)(ap + k0 + 16);    // K = kA+16 .. kA+23
    b.q[0] = *(const uint4*)(bp + k0);         // K = kB .. kB+7
    b.q[1] = *(const uint4*)(bp + k0 + 16);    // K = kB+16 .. kB+23
    acc = __builtin_amdgcn_wmma_f32_16x16x32_bf16(
        /*neg_a=*/false, a.v, /*neg_b=*/false, b.v,
        /*c_mod=*/(short)0, acc, /*reuse_a=*/false, /*reuse_b=*/false);
  }

  // D layout: lane L holds col N=L%16, rows M = 8*(L/16)+i, i=0..7.
  int n     = tn * 16 + (lane & 15);
  int mbase = tm * 16 + (lane >> 4) * 8;
  float bv = bias ? bias[n] : 0.0f;
  #pragma unroll
  for (int i = 0; i < 8; ++i)
    C[(size_t)(mbase + i) * ldc + n] = acc[i] + bv;
}

// ---------------------------------------------------------------------------
// Persistent sequential LSTM layer (single batch row).
// 64 WGs x 256 threads. WG w owns h-columns [16w, 16w+16) i.e. z-columns
// {g*1024 + 16w + j : g<4, j<16}. Its 64x1024 bf16 slice of WhT lives in LDS
// (128 KB of the WGP's 320 KB) for all 512 steps. Steps are separated by a
// device-wide generation barrier (atomics + s_sleep backoff).
__global__ __launch_bounds__(256) void lstm_seq(
    const uint16_t* __restrict__ WhT,    // GATES x HDIM bf16 (row = global z-col)
    const float*    __restrict__ Zx,     // TSTEPS x GATES precomputed x-proj (+bias)
    const float*    __restrict__ c_init, // HDIM f32 (slice of state row 63)
    uint16_t*       __restrict__ Hbuf,   // (TSTEPS+1) x HDIM bf16; row 0 = h_init
    unsigned* cnt, unsigned* gen, int T) {
  __shared__ uint16_t wlds[ZCOLS * WPAD];   // ~129 KB
  __shared__ float    hl[HDIM];             // 4 KB
  __shared__ float    zsum[ZCOLS * 4];      // 1 KB
  __shared__ float    c_lds[16];

  const int tid = threadIdx.x;
  const int w   = blockIdx.x;

  // Preload recurrent weight slice (coalesced along K).
  for (int i = tid; i < ZCOLS * HDIM; i += 256) {
    int l = i >> 10;                // local z-col 0..63
    int k = i & (HDIM - 1);
    int zc = ((l >> 4) << 10) + (w << 4) + (l & 15);   // global z-col
    wlds[l * WPAD + k] = WhT[(size_t)zc * HDIM + k];
  }
  if (tid < 16) c_lds[tid] = c_init[(w << 4) + tid];
  __syncthreads();

  const int col = tid & 63;                  // local z-col
  const int q   = tid >> 6;                  // K-quarter 0..3
  const uint16_t* wrow = &wlds[col * WPAD + q * 256];
  const float*    hq   = &hl[q * 256];

  for (int t = 0; t < T; ++t) {
    // h_{t-1} (row t of Hbuf) -> LDS f32
    const uint32_t* hrow = (const uint32_t*)(Hbuf + (size_t)t * HDIM);
    for (int i = tid; i < HDIM / 2; i += 256) {
      uint32_t u = hrow[i];
      hl[2 * i]     = bflo(u);
      hl[2 * i + 1] = bfhi(u);
    }
    __syncthreads();

    // 64 dot products of length 1024, 4 threads per column.
    float s = 0.0f;
    #pragma unroll 4
    for (int k = 0; k < 256; k += 8) {
      uint4 wq = *(const uint4*)(wrow + k);
      s += bflo(wq.x) * hq[k]     + bfhi(wq.x) * hq[k + 1];
      s += bflo(wq.y) * hq[k + 2] + bfhi(wq.y) * hq[k + 3];
      s += bflo(wq.z) * hq[k + 4] + bfhi(wq.z) * hq[k + 5];
      s += bflo(wq.w) * hq[k + 6] + bfhi(wq.w) * hq[k + 7];
    }
    zsum[col * 4 + q] = s;
    __syncthreads();

    // Gate math for the 16 owned h-columns (fp32).
    if (tid < 16) {
      const float* zx = Zx + (size_t)t * GATES + (w << 4) + tid;
      float z[4];
      #pragma unroll
      for (int g = 0; g < 4; ++g) {
        int l = g * 16 + tid;
        z[g] = zx[g << 10] + zsum[l * 4] + zsum[l * 4 + 1] + zsum[l * 4 + 2] + zsum[l * 4 + 3];
      }
      float c  = c_lds[tid];
      float c2 = sigmoidf(z[2] + 1.0f) * c + sigmoidf(z[0]) * tanhf(z[1]);  // f gate + FORGET_BIAS
      float h2 = sigmoidf(z[3]) * tanhf(c2);
      c_lds[tid] = c2;
      Hbuf[(size_t)(t + 1) * HDIM + (w << 4) + tid] = f2bf(h2);
    }

    // Device-wide barrier so step t+1 sees every WG's h write.
    __threadfence();
    __syncthreads();
    if (tid == 0) {
      unsigned prev = __hip_atomic_fetch_add(cnt, 1u, __ATOMIC_ACQ_REL, __HIP_MEMORY_SCOPE_AGENT);
      if (prev == (unsigned)(NWG - 1)) {
        __hip_atomic_store(cnt, 0u, __ATOMIC_RELAXED, __HIP_MEMORY_SCOPE_AGENT);
        __hip_atomic_fetch_add(gen, 1u, __ATOMIC_RELEASE, __HIP_MEMORY_SCOPE_AGENT);
      } else {
        while (__hip_atomic_load(gen, __ATOMIC_ACQUIRE, __HIP_MEMORY_SCOPE_AGENT) < (unsigned)(t + 1)) {
          __builtin_amdgcn_s_sleep(2);
        }
      }
    }
    __syncthreads();
    __threadfence();   // acquire: invalidate before reading peers' h next step
  }
}

// ---------------------------------------------------------------------------
extern "C" void kernel_launch(void* const* d_in, const int* in_sizes, int n_in,
                              void* d_out, int out_size, void* d_ws, size_t ws_size,
                              hipStream_t stream) {
  const float* X  = (const float*)d_in[0];   // (64, 512, 512)
  const float* st = (const float*)d_in[1];   // (64, 4096)
  const float* W0 = (const float*)d_in[2];   // (1536, 4096)
  const float* b0 = (const float*)d_in[3];   // (4096,)
  const float* W1 = (const float*)d_in[4];   // (2048, 4096)
  const float* b1 = (const float*)d_in[5];   // (4096,)
  const float* Wd = (const float*)d_in[6];   // (1024, 512)
  const float* bd = (const float*)d_in[7];   // (512,)
  float* out = (float*)d_out;                // (512, 512)

  char* p = (char*)d_ws;
  auto carve = [&](size_t bytes) { char* r = p; p += (bytes + 255) & ~(size_t)255; return r; };
  uint16_t* W0xT  = (uint16_t*)carve((size_t)GATES * INSZ * 2);   // 4096 x 512
  uint16_t* W0hT  = (uint16_t*)carve((size_t)GATES * HDIM * 2);   // 4096 x 1024
  uint16_t* W1xT  = (uint16_t*)carve((size_t)GATES * HDIM * 2);
  uint16_t* W1hT  = (uint16_t*)carve((size_t)GATES * HDIM * 2);
  uint16_t* WdT   = (uint16_t*)carve((size_t)OUTSZ * HDIM * 2);   // 512 x 1024
  uint16_t* X63b  = (uint16_t*)carve((size_t)TSTEPS * INSZ * 2);
  uint16_t* Hbuf0 = (uint16_t*)carve((size_t)(TSTEPS + 1) * HDIM * 2);
  uint16_t* Hbuf1 = (uint16_t*)carve((size_t)(TSTEPS + 1) * HDIM * 2);
  float*    Z0x   = (float*)carve((size_t)TSTEPS * GATES * 4);
  float*    Z1x   = (float*)carve((size_t)TSTEPS * GATES * 4);
  unsigned* ctrs  = (unsigned*)carve(256);

  const float* state63 = st + (size_t)63 * 4 * HDIM;
  const float* X63     = X + (size_t)63 * TSTEPS * INSZ;

  auto blk = [](size_t n) { return (unsigned)((n + 255) / 256); };

  // Weight prep (bf16, transposed so WMMA B-operand K-runs are contiguous).
  hipLaunchKernelGGL(cast_transpose, dim3(blk((size_t)INSZ * GATES)), dim3(256), 0, stream,
                     W0, W0xT, INSZ, GATES);
  hipLaunchKernelGGL(cast_transpose, dim3(blk((size_t)HDIM * GATES)), dim3(256), 0, stream,
                     W0 + (size_t)INSZ * GATES, W0hT, HDIM, GATES);
  hipLaunchKernelGGL(cast_transpose, dim3(blk((size_t)HDIM * GATES)), dim3(256), 0, stream,
                     W1, W1xT, HDIM, GATES);
  hipLaunchKernelGGL(cast_transpose, dim3(blk((size_t)HDIM * GATES)), dim3(256), 0, stream,
                     W1 + (size_t)HDIM * GATES, W1hT, HDIM, GATES);
  hipLaunchKernelGGL(cast_transpose, dim3(blk((size_t)HDIM * OUTSZ)), dim3(256), 0, stream,
                     Wd, WdT, HDIM, OUTSZ);
  hipLaunchKernelGGL(cast_copy, dim3(blk((size_t)TSTEPS * INSZ)), dim3(256), 0, stream,
                     X63, X63b, TSTEPS * INSZ);
  hipLaunchKernelGGL(init_misc, dim3(4), dim3(256), 0, stream, state63, Hbuf0, Hbuf1, ctrs);

  // Z0x = X63 @ W0[:512] + b0   (512 x 4096)
  {
    int Mt = TSTEPS / 16, Nt = GATES / 16;
    hipLaunchKernelGGL(gemm_bf16, dim3((Mt * Nt + 3) / 4), dim3(128), 0, stream,
                       X63b, INSZ, W0xT, INSZ, b0, Z0x, GATES, Mt, Nt, INSZ);
  }
  // Layer 0 recurrence.
  hipLaunchKernelGGL(lstm_seq, dim3(NWG), dim3(256), 0, stream,
                     W0hT, Z0x, state63 + 0, Hbuf0, ctrs + 0, ctrs + 1, TSTEPS);
  // Z1x = H0 @ W1[:1024] + b1   (512 x 4096)
  {
    int Mt = TSTEPS / 16, Nt = GATES / 16;
    hipLaunchKernelGGL(gemm_bf16, dim3((Mt * Nt + 3) / 4), dim3(128), 0, stream,
                       Hbuf0 + HDIM, HDIM, W1xT, HDIM, b1, Z1x, GATES, Mt, Nt, HDIM);
  }
  // Layer 1 recurrence (c1 = state[63, 2H:3H]).
  hipLaunchKernelGGL(lstm_seq, dim3(NWG), dim3(256), 0, stream,
                     W1hT, Z1x, state63 + 2 * HDIM, Hbuf1, ctrs + 2, ctrs + 3, TSTEPS);
  // logits = H1 @ Wd + bd   (512 x 512)
  {
    int Mt = TSTEPS / 16, Nt = OUTSZ / 16;
    hipLaunchKernelGGL(gemm_bf16, dim3((Mt * Nt + 3) / 4), dim3(128), 0, stream,
                       Hbuf1 + HDIM, HDIM, WdT, HDIM, bd, out, OUTSZ, Mt, Nt, HDIM);
  }
}
